// SAGE_3547642986708
// MI455X (gfx1250) — matrix operand order repaired and checked by reading, not compiled, not run on hardware
//
#include <hip/hip_runtime.h>

// ---------------------------------------------------------------------------
// GraphSAGE 3-layer minibatch forward for MI455X (gfx1250, wave32, WMMA).
// Memory-bound on the layer-1 edge gather (~405MB random reads); GEMMs use
// v_wmma_f32_16x16x32_bf16 (bf16 in, f32 accumulate).
// ---------------------------------------------------------------------------

typedef __attribute__((ext_vector_type(16))) __bf16 v16bf;
typedef __attribute__((ext_vector_type(8)))  float  v8f;

#define N0v   1000000
#define N1v   67584
#define N2v   6144
#define N3v   1024
#define FINv  100
#define HIDv  256
#define OUTv  47
#define FAN1v 15
#define FAN2v 10
#define FAN3v 5

static __device__ __forceinline__ unsigned short f32_to_bf16(float f) {
    union { float f; unsigned u; } c; c.f = f;
    unsigned u = c.u;
    u = u + 0x7FFFu + ((u >> 16) & 1u);   // round-to-nearest-even
    return (unsigned short)(u >> 16);
}

// ---------------------------------------------------------------------------
// Pack fp32 weights [K][N] into bf16 WMMA-B fragment order:
//   P[((nt*ktiles + kt)*32 + lane)*16 + e] = W[kt*32 + (lane>>4)*16 + e][nt*16 + (lane&15)]
// zero-padded to Kp x Np.  Per-lane B fragment load becomes 2 coalesced uint4.
// ---------------------------------------------------------------------------
__global__ void pack_w_kernel(const float* __restrict__ W, unsigned short* __restrict__ P,
                              int K, int N, int Kp, int Np) {
    int tid = blockIdx.x * blockDim.x + threadIdx.x;
    int total = Kp * Np;
    if (tid >= total) return;
    int ktiles = Kp >> 5;
    int e    = tid & 15;
    int lane = (tid >> 4) & 31;
    int t    = tid >> 9;                 // 32 lanes * 16 elems = 512 per tile
    int kt   = t % ktiles;
    int nt   = t / ktiles;
    int k = (kt << 5) + ((lane >> 4) << 4) + e;
    int n = (nt << 4) + (lane & 15);
    float v = (k < K && n < N) ? W[(long long)k * N + n] : 0.0f;
    P[tid] = f32_to_bf16(v);
}

// ---------------------------------------------------------------------------
// Mean aggregation over `fan` contiguous edges per dst node (dst=repeat(arange)).
// One block per dst node, one thread per feature; each neighbor-row read is
// coalesced across the feature lanes.  Prefetch the next neighbor's row.
// ---------------------------------------------------------------------------
__global__ void sage_mean_agg(const float* __restrict__ xsrc, const int* __restrict__ src,
                              float* __restrict__ agg, int feat, int fan, float inv_fan) {
    int node = blockIdx.x;
    int f = threadIdx.x;
    if (f >= feat) return;
    const int* s = src + (long long)node * fan;
    float acc = 0.0f;
    for (int j = 0; j < fan; ++j) {
        long long r = (long long)s[j];
        if (j + 1 < fan)
            __builtin_prefetch(xsrc + (long long)s[j + 1] * feat + f, 0, 0);
        acc += xsrc[r * feat + f];
    }
    agg[(long long)node * feat + f] = acc * inv_fan;
}

// ---------------------------------------------------------------------------
// Fused dual GEMM:  C[16 rows] = relu?( A1@W1 + A2@W2 + bias )
//   A1,A2: fp32, row-major, real K cols (staged to LDS as bf16, padded to Kp)
//   P1,P2: bf16 fragment-packed weights (Kp x Np)
// Block = 256 threads (8 waves); block handles 16 rows x Np cols; each wave
// owns ntile(s) of 16 columns.  M is a multiple of 16 for every layer.
// ---------------------------------------------------------------------------
#define GT 256
__global__ void sage_gemm_wmma(const float* __restrict__ A1, int lda1,
                               const float* __restrict__ A2, int lda2,
                               const unsigned short* __restrict__ P1,
                               const unsigned short* __restrict__ P2,
                               const float* __restrict__ bias,
                               float* __restrict__ Cout, int ldc,
                               int K, int Kp, int Nreal, int Np, int relu_flag) {
    extern __shared__ uint4 lds_raw[];
    unsigned short* A1t = (unsigned short*)lds_raw;       // [16][Kp] bf16
    unsigned short* A2t = A1t + 16 * Kp;                  // [16][Kp] bf16

    const int row0 = blockIdx.x << 4;
    const int tid  = threadIdx.x;
    const int kmask  = Kp - 1;            // Kp is a power of two (128/256)
    const int kshift = 31 - __clz(Kp);

    // Stage both A tiles into LDS as bf16 (zero-pad K..Kp-1).
    for (int idx = tid; idx < (Kp << 4); idx += GT) {
        int r = idx >> kshift;
        int c = idx & kmask;
        float v1 = 0.0f, v2 = 0.0f;
        if (c < K) {
            v1 = A1[(long long)(row0 + r) * lda1 + c];
            v2 = A2[(long long)(row0 + r) * lda2 + c];
        }
        A1t[idx] = f32_to_bf16(v1);
        A2t[idx] = f32_to_bf16(v2);
    }
    __syncthreads();

    const int lane  = tid & 31;
    const int wave  = tid >> 5;
    const int ktiles = Kp >> 5;
    const int ntiles = Np >> 4;
    const int mrow  = lane & 15;
    const int khalf = (lane >> 4) << 4;   // 0 or 16: K sub-block per half-wave

    union Frag { uint4 q[2]; v16bf v; };

    for (int nt = wave; nt < ntiles; nt += (GT >> 5)) {
        v8f acc = {};
        const unsigned short* p1 = P1 + ((long long)nt * ktiles << 9) + (lane << 4);
        const unsigned short* p2 = P2 + ((long long)nt * ktiles << 9) + (lane << 4);

        // pass 1: A1 @ W1
        for (int kt = 0; kt < ktiles; ++kt) {
            Frag a, b;
            const unsigned short* ab = A1t + mrow * Kp + (kt << 5) + khalf;
            a.q[0] = *(const uint4*)(ab);          // K = k0 .. k0+7
            a.q[1] = *(const uint4*)(ab + 8);      // K = k0+8 .. k0+15
            b.q[0] = *(const uint4*)(p1 + (kt << 9));
            b.q[1] = *(const uint4*)(p1 + (kt << 9) + 8);
            acc = __builtin_amdgcn_wmma_f32_16x16x32_bf16(
                      false, a.v, false, b.v, (short)0, acc, false, false);
        }
        // pass 2: A2 @ W2 (accumulate)
        for (int kt = 0; kt < ktiles; ++kt) {
            Frag a, b;
            const unsigned short* ab = A2t + mrow * Kp + (kt << 5) + khalf;
            a.q[0] = *(const uint4*)(ab);
            a.q[1] = *(const uint4*)(ab + 8);
            b.q[0] = *(const uint4*)(p2 + (kt << 9));
            b.q[1] = *(const uint4*)(p2 + (kt << 9) + 8);
            acc = __builtin_amdgcn_wmma_f32_16x16x32_bf16(
                      false, a.v, false, b.v, (short)0, acc, false, false);
        }

        // Epilogue: C/D layout -> n = lane&15, m = r + 8*(lane>>4)
        const int n = (nt << 4) + (lane & 15);
        const int mbase = (lane >> 4) << 3;
        const float bv = (n < Nreal) ? bias[n] : 0.0f;
        for (int r = 0; r < 8; ++r) {
            float v = acc[r] + bv;
            if (relu_flag) v = fmaxf(v, 0.0f);
            Cout[(long long)(row0 + mbase + r) * ldc + n] = v;
        }
    }
}

// ---------------------------------------------------------------------------
// log_softmax over 47 classes (logits stored with stride 48, pad col ignored)
// ---------------------------------------------------------------------------
__global__ void log_softmax_47(const float* __restrict__ logits, float* __restrict__ out) {
    int row = blockIdx.x * blockDim.x + threadIdx.x;
    if (row >= N3v) return;
    const float* l = logits + row * 48;
    float m = -3.402823466e38f;
    for (int j = 0; j < OUTv; ++j) m = fmaxf(m, l[j]);
    float s = 0.0f;
    for (int j = 0; j < OUTv; ++j) s += __expf(l[j] - m);
    float lse = m + __logf(s);
    for (int j = 0; j < OUTv; ++j) out[(long long)row * OUTv + j] = l[j] - lse;
}

// ---------------------------------------------------------------------------
extern "C" void kernel_launch(void* const* d_in, const int* in_sizes, int n_in,
                              void* d_out, int out_size, void* d_ws, size_t ws_size,
                              hipStream_t stream) {
    (void)in_sizes; (void)n_in; (void)out_size; (void)ws_size;

    const float* x    = (const float*)d_in[0];
    const float* W_l1 = (const float*)d_in[1];
    const float* W_r1 = (const float*)d_in[2];
    const float* b1   = (const float*)d_in[3];
    const float* W_l2 = (const float*)d_in[4];
    const float* W_r2 = (const float*)d_in[5];
    const float* b2   = (const float*)d_in[6];
    const float* W_l3 = (const float*)d_in[7];
    const float* W_r3 = (const float*)d_in[8];
    const float* b3   = (const float*)d_in[9];
    const int*   src1 = (const int*)d_in[10];
    const int*   src2 = (const int*)d_in[12];
    const int*   src3 = (const int*)d_in[14];
    float* out = (float*)d_out;

    // ---- workspace carve-up (256B aligned) ----
    char* ws = (char*)d_ws;
    size_t off = 0;
    auto take = [&](size_t bytes) -> char* {
        char* p = ws + off;
        off += (bytes + 255) & ~(size_t)255;
        return p;
    };
    unsigned short* P1l = (unsigned short*)take((size_t)128 * 256 * 2);
    unsigned short* P1r = (unsigned short*)take((size_t)128 * 256 * 2);
    unsigned short* P2l = (unsigned short*)take((size_t)256 * 256 * 2);
    unsigned short* P2r = (unsigned short*)take((size_t)256 * 256 * 2);
    unsigned short* P3l = (unsigned short*)take((size_t)256 * 48 * 2);
    unsigned short* P3r = (unsigned short*)take((size_t)256 * 48 * 2);
    float* agg1   = (float*)take((size_t)N1v * FINv * 4);
    float* h1     = (float*)take((size_t)N1v * HIDv * 4);
    float* agg2   = (float*)take((size_t)N2v * HIDv * 4);
    float* h2     = (float*)take((size_t)N2v * HIDv * 4);
    float* agg3   = (float*)take((size_t)N3v * HIDv * 4);
    float* logits = (float*)take((size_t)N3v * 48 * 4);

    // ---- weight packing (tiny) ----
    pack_w_kernel<<<(128 * 256 + 255) / 256, 256, 0, stream>>>(W_l1, P1l, FINv, HIDv, 128, 256);
    pack_w_kernel<<<(128 * 256 + 255) / 256, 256, 0, stream>>>(W_r1, P1r, FINv, HIDv, 128, 256);
    pack_w_kernel<<<(256 * 256 + 255) / 256, 256, 0, stream>>>(W_l2, P2l, HIDv, HIDv, 256, 256);
    pack_w_kernel<<<(256 * 256 + 255) / 256, 256, 0, stream>>>(W_r2, P2r, HIDv, HIDv, 256, 256);
    pack_w_kernel<<<(256 * 48 + 255) / 256, 256, 0, stream>>>(W_l3, P3l, HIDv, OUTv, 256, 48);
    pack_w_kernel<<<(256 * 48 + 255) / 256, 256, 0, stream>>>(W_r3, P3r, HIDv, OUTv, 256, 48);

    // ---- layer 1 ----
    sage_mean_agg<<<N1v, 128, 0, stream>>>(x, src1, agg1, FINv, FAN1v, 1.0f / FAN1v);
    sage_gemm_wmma<<<N1v / 16, GT, 2 * 16 * 128 * sizeof(unsigned short), stream>>>(
        agg1, FINv, x, FINv, P1l, P1r, b1, h1, HIDv,
        FINv, 128, HIDv, 256, 1);

    // ---- layer 2 ----
    sage_mean_agg<<<N2v, 256, 0, stream>>>(h1, src2, agg2, HIDv, FAN2v, 1.0f / FAN2v);
    sage_gemm_wmma<<<N2v / 16, GT, 2 * 16 * 256 * sizeof(unsigned short), stream>>>(
        agg2, HIDv, h1, HIDv, P2l, P2r, b2, h2, HIDv,
        HIDv, 256, HIDv, 256, 1);

    // ---- layer 3 ----
    sage_mean_agg<<<N3v, 256, 0, stream>>>(h2, src3, agg3, HIDv, FAN3v, 1.0f / FAN3v);
    sage_gemm_wmma<<<N3v / 16, GT, 2 * 16 * 256 * sizeof(unsigned short), stream>>>(
        agg3, HIDv, h2, HIDv, P3l, P3r, b3, logits, 48,
        HIDv, 256, OUTv, 48, 0);

    // ---- log-softmax ----
    log_softmax_47<<<(N3v + 255) / 256, 256, 0, stream>>>(logits, out);
}
